// SemanticBasisV4_21328807592426
// MI455X (gfx1250) — matrix operand order repaired and checked by reading, not compiled, not run on hardware
//
#include <hip/hip_runtime.h>
#include <hip/hip_bf16.h>

// ---------------------------------------------------------------------------
// SemanticBasisV4 for gfx1250 (MI455X): bf16 WMMA GEMM pipeline with
// double-buffered GLOBAL_LOAD_ASYNC_TO_LDS_B128 tile staging.
//   E[0:1024]      = U @ B
//   E[1024:V]      = (gelu(gelu(phi@W1+b1)@W2+b2)@W3+b3) * tail_gain
//   latent         = E[old_to_new[tokens]]            (perms are inverses)
//   logits[i][v]   = (latent[i] . E[old_to_new[v]]) * exp(log_alpha) + out_bias[v]
// Weights are pre-transposed to N x K so A and B tiles are both contiguous
// 32B/row copies -> async-to-LDS friendly, no transpose in the hot loop.
// ---------------------------------------------------------------------------

typedef __attribute__((ext_vector_type(16))) __bf16 v16bf;
typedef __attribute__((ext_vector_type(8)))  float  v8f;

static constexpr int kV     = 50257;
static constexpr int kD     = 1024;
static constexpr int kHot   = 1024;
static constexpr int kRHot  = 768;
static constexpr int kHid   = 1024;
static constexpr int kNF    = 48;
static constexpr int kInDim = 2 * kNF + 8;     // 104
static constexpr int kKPad  = 128;             // padded K for phi@W1
static constexpr int kMC    = kV - kHot;       // 49233 cold rows
static constexpr int kMCP   = 49280;           // 385 * 128 (padded)
static constexpr int kVPad  = kHot + kMCP;     // 50304 E rows (padded)
static constexpr int kTok   = 2048;            // 2 * 1024 tokens

static constexpr int kLdsStride = 40;          // ushorts per LDS row (80B, 16B-mult)
static constexpr int kBufBytes  = 128 * kLdsStride * 2;   // one tile buffer

__device__ __forceinline__ unsigned short f32_to_bf16(float f) {
  unsigned int u = __builtin_bit_cast(unsigned int, f);
  unsigned int r = (u + 0x7fffu + ((u >> 16) & 1u)) >> 16;   // RNE
  return (unsigned short)r;
}
__device__ __forceinline__ __bf16 us_to_bf(unsigned short u) {
  union { unsigned short s; __bf16 h; } c; c.s = u; return c.h;
}
__device__ __forceinline__ float gelu_exact(float x) {
  return 0.5f * x * (1.0f + erff(x * 0.70710678118654752f));
}
// One 16-byte global->LDS async copy (ASYNCcnt-tracked, bypasses VGPRs).
__device__ __forceinline__ void async_cp16(unsigned lds_off,
                                           const unsigned short* g) {
  unsigned long long ga = (unsigned long long)(size_t)g;
  asm volatile("global_load_async_to_lds_b128 %0, %1, off"
               :: "v"(lds_off), "v"(ga) : "memory");
}

// ---------------------------- prep kernels ----------------------------------

__global__ void k_f32_to_bf16(const float* __restrict__ src,
                              unsigned short* __restrict__ dst, int n) {
  int i = blockIdx.x * blockDim.x + threadIdx.x;
  if (i < n) dst[i] = f32_to_bf16(src[i]);
}

// dst[n*K + k] = bf16(src[k*N + n])   (weights K x N  ->  N x K)
__global__ void k_cvt_transpose(const float* __restrict__ src,
                                unsigned short* __restrict__ dst, int K, int N) {
  int i = blockIdx.x * blockDim.x + threadIdx.x;
  if (i >= K * N) return;
  int k = i / N, n = i - k * N;
  dst[(size_t)n * K + k] = f32_to_bf16(src[i]);
}

// W1 (104 x 1024) -> W1T (1024 x 128), K padded with zeros.
__global__ void k_w1_pad_t(const float* __restrict__ W1,
                           unsigned short* __restrict__ dst) {
  int i = blockIdx.x * blockDim.x + threadIdx.x;   // over 1024*128
  if (i >= kHid * kKPad) return;
  int n = i / kKPad, r = i - n * kKPad;
  dst[i] = (r < kInDim) ? f32_to_bf16(W1[r * kHid + n]) : (unsigned short)0;
}

// Fourier features of cold ids, bf16, K padded 104 -> 128 with zeros.
__global__ void k_phi(const int* __restrict__ n2o, const float* __restrict__ freqs,
                      unsigned short* __restrict__ phi) {
  int r = blockIdx.x * blockDim.x + threadIdx.x;
  if (r >= kMCP) return;
  unsigned short* row = phi + (size_t)r * kKPad;
  if (r < kMC) {
    int id = n2o[kHot + r];
    float x = (float)id / (float)kV;
    for (int f = 0; f < kNF; ++f) {
      float s, c;
      sincosf(x * freqs[f], &s, &c);
      row[f]       = f32_to_bf16(s);
      row[kNF + f] = f32_to_bf16(c);
    }
    for (int k = 0; k < 8; ++k)
      row[2 * kNF + k] = f32_to_bf16((float)((id >> (2 * k)) & 3) * (1.0f / 3.0f));
    for (int j = kInDim; j < kKPad; ++j) row[j] = 0;
  } else {
    for (int j = 0; j < kKPad; ++j) row[j] = 0;
  }
}

// latent[i] = E[old_to_new[tokens[i]]]  (bf16 row copy)
__global__ void k_latent(const int* __restrict__ tokens, const int* __restrict__ o2n,
                         const unsigned short* __restrict__ E,
                         unsigned short* __restrict__ L) {
  int i = blockIdx.x;
  int nt = o2n[tokens[i]];
  const unsigned int* src = (const unsigned int*)(E + (size_t)nt * kD);
  unsigned int* dst = (unsigned int*)(L + (size_t)i * kD);
  for (int j = threadIdx.x; j < kD / 2; j += blockDim.x) dst[j] = src[j];
}

// ------------------------------- WMMA GEMM ---------------------------------
// A: M x K bf16 row-major.  BT: N x K bf16 row-major (pre-transposed / E rows).
// GATHER==0: Cbf = act(A @ BT^T + bias) * scale   (bf16 out, M,N mult of 128)
// GATHER==1: Cf  = (A @ BT[map[n]]^T) * exp(alpha) + bias[n]  (f32, n < Nreal)
// Block 128x128, K-step 32, 8 waves x (32x64) = 2x4 WMMA accs per wave.
// Tiles staged with double-buffered global_load_async_to_lds_b128.
template<int GATHER>
__global__ __launch_bounds__(256) void k_gemm(
    const unsigned short* __restrict__ A,
    const unsigned short* __restrict__ BT,
    unsigned short* __restrict__ Cbf,
    float* __restrict__ Cf,
    int M, int N, int K, int Nreal,
    const int* __restrict__ map,
    const float* __restrict__ bias, int act,
    const float* __restrict__ scale_ptr,
    const float* __restrict__ alpha_ptr)
{
  __shared__ alignas(16) unsigned short As[2][128][kLdsStride]; // [m][k]
  __shared__ alignas(16) unsigned short Bs[2][128][kLdsStride]; // [n][k]

  const int tid  = threadIdx.x;
  const int lane = tid & 31;
  const int wid  = tid >> 5;
  const int wm_base = (wid & 3) * 32;
  const int wn_base = (wid >> 2) * 64;
  const int m0 = blockIdx.y * 128;
  const int n0 = blockIdx.x * 128;
  const int halfsel = (lane < 16) ? 0 : 1;
  const int lrow = lane & 15;

  // per-thread tile-copy assignment: row r, 16-element (32B) chunk cb
  const int r  = tid >> 1;
  const int cb = (tid & 1) * 16;
  const unsigned short* Agb = A + (size_t)(m0 + r) * K + cb;
  int brow;
  if (GATHER) {
    int gcol = n0 + r;
    brow = (gcol < Nreal) ? map[gcol] : 0;
  } else {
    brow = n0 + r;
  }
  const unsigned short* Bgb = BT + (size_t)brow * K + cb;
  const unsigned laA = (unsigned)(size_t)&As[0][r][cb];
  const unsigned laB = (unsigned)(size_t)&Bs[0][r][cb];

  auto issue = [&](int buf, int k0) {
    unsigned da = laA + (unsigned)buf * kBufBytes;
    unsigned db = laB + (unsigned)buf * kBufBytes;
    async_cp16(da,      Agb + k0);
    async_cp16(da + 16, Agb + k0 + 8);
    async_cp16(db,      Bgb + k0);
    async_cp16(db + 16, Bgb + k0 + 8);
  };

  v8f acc[2][4];
#pragma unroll
  for (int i = 0; i < 2; ++i)
#pragma unroll
    for (int j = 0; j < 4; ++j) {
      v8f z = {0.f, 0.f, 0.f, 0.f, 0.f, 0.f, 0.f, 0.f};
      acc[i][j] = z;
    }

  const int nk = K >> 5;
  issue(0, 0);
  for (int it = 0; it < nk; ++it) {
    const int cur = it & 1;
    if (it + 1 < nk) {
      issue(cur ^ 1, (it + 1) << 5);
      asm volatile("s_wait_asynccnt 0x4" ::: "memory");  // current buf done
    } else {
      asm volatile("s_wait_asynccnt 0x0" ::: "memory");
    }
    __syncthreads();

    // ---- fragments in contiguous k order (k = 16*half + s), 8 WMMAs ----
    v16bf afrag[2];
#pragma unroll
    for (int wm = 0; wm < 2; ++wm) {
      const unsigned short* p = &As[cur][wm_base + wm * 16 + lrow][halfsel * 16];
#pragma unroll
      for (int q = 0; q < 4; ++q) {            // K = base + {0..7}
        unsigned int pr = *(const unsigned int*)(p + 2 * q);
        afrag[wm][2 * q]     = us_to_bf((unsigned short)pr);
        afrag[wm][2 * q + 1] = us_to_bf((unsigned short)(pr >> 16));
      }
#pragma unroll
      for (int q = 0; q < 4; ++q) {            // K = base + 8 + {0..7}
        unsigned int pr = *(const unsigned int*)(p + 8 + 2 * q);
        afrag[wm][8 + 2 * q]     = us_to_bf((unsigned short)pr);
        afrag[wm][8 + 2 * q + 1] = us_to_bf((unsigned short)(pr >> 16));
      }
    }
#pragma unroll
    for (int wn = 0; wn < 4; ++wn) {
      v16bf bfrag;
      const unsigned short* p = &Bs[cur][wn_base + wn * 16 + lrow][halfsel * 16];
#pragma unroll
      for (int q = 0; q < 8; ++q) {
        unsigned int pr = *(const unsigned int*)(p + 2 * q);
        bfrag[2 * q]     = us_to_bf((unsigned short)pr);
        bfrag[2 * q + 1] = us_to_bf((unsigned short)(pr >> 16));
      }
#pragma unroll
      for (int wm = 0; wm < 2; ++wm) {
        acc[wm][wn] = __builtin_amdgcn_wmma_f32_16x16x32_bf16(
            false, afrag[wm], false, bfrag, (short)0, acc[wm][wn], false, false);
      }
    }
    __syncthreads();   // all waves done reading `cur` before it is refilled
  }

  // ---- epilogue (C layout: VGPR v -> M=v / v+8; lane&15 -> N) ----
  float sc = 1.0f, alpha = 1.0f;
  if (!GATHER && scale_ptr) sc = *scale_ptr;
  if (GATHER) alpha = expf(*alpha_ptr);
#pragma unroll
  for (int wm = 0; wm < 2; ++wm) {
    int rb = m0 + wm_base + wm * 16 + halfsel * 8;
#pragma unroll
    for (int wn = 0; wn < 4; ++wn) {
      int col = n0 + wn_base + wn * 16 + lrow;
      if (GATHER) {
        if (col < Nreal) {
          float ob = bias[col];
#pragma unroll
          for (int v = 0; v < 8; ++v)
            Cf[(size_t)(rb + v) * Nreal + col] = acc[wm][wn][v] * alpha + ob;
        }
      } else {
        float bs = bias ? bias[col] : 0.0f;
#pragma unroll
        for (int v = 0; v < 8; ++v) {
          float x = acc[wm][wn][v] + bs;
          if (act) x = gelu_exact(x);
          Cbf[(size_t)(rb + v) * N + col] = f32_to_bf16(x * sc);
        }
      }
    }
  }
}

// --------------------------------- launch ----------------------------------

extern "C" void kernel_launch(void* const* d_in, const int* in_sizes, int n_in,
                              void* d_out, int out_size, void* d_ws, size_t ws_size,
                              hipStream_t stream) {
  const int*   tokens = (const int*)  d_in[0];
  const int*   o2n    = (const int*)  d_in[1];
  const int*   n2o    = (const int*)  d_in[2];
  const float* freqs  = (const float*)d_in[3];
  const float* U      = (const float*)d_in[4];
  const float* Bm     = (const float*)d_in[5];
  const float* W1     = (const float*)d_in[6];
  const float* b1     = (const float*)d_in[7];
  const float* W2     = (const float*)d_in[8];
  const float* b2     = (const float*)d_in[9];
  const float* W3     = (const float*)d_in[10];
  const float* b3     = (const float*)d_in[11];
  const float* gain   = (const float*)d_in[12];
  const float* lalpha = (const float*)d_in[13];
  const float* obias  = (const float*)d_in[14];
  float* out = (float*)d_out;

  // workspace carve-out (~326 MB total)
  char* ws = (char*)d_ws;
  size_t off = 0;
  auto carve = [&](size_t bytes) -> void* {
    void* p = ws + off;
    off += (bytes + 255) & ~(size_t)255;
    return p;
  };
  unsigned short* Ubf  = (unsigned short*)carve((size_t)kHot * kRHot * 2);  // M x K
  unsigned short* BmT  = (unsigned short*)carve((size_t)kD   * kRHot * 2);  // N x K
  unsigned short* W1T  = (unsigned short*)carve((size_t)kHid * kKPad * 2);  // N x K
  unsigned short* W2T  = (unsigned short*)carve((size_t)kHid * kHid  * 2);  // N x K
  unsigned short* W3T  = (unsigned short*)carve((size_t)kD   * kHid  * 2);  // N x K
  unsigned short* Phi  = (unsigned short*)carve((size_t)kMCP * kKPad * 2);
  unsigned short* H1   = (unsigned short*)carve((size_t)kMCP * kHid  * 2);
  unsigned short* H2   = (unsigned short*)carve((size_t)kMCP * kHid  * 2);
  unsigned short* Ebf  = (unsigned short*)carve((size_t)kVPad * kD   * 2);
  unsigned short* Lat  = (unsigned short*)carve((size_t)kTok * kD    * 2);

  k_f32_to_bf16<<<(kHot * kRHot + 255) / 256, 256, 0, stream>>>(U, Ubf, kHot * kRHot);
  k_cvt_transpose<<<(kRHot * kD + 255) / 256, 256, 0, stream>>>(Bm, BmT, kRHot, kD);
  k_cvt_transpose<<<(kHid * kHid + 255) / 256, 256, 0, stream>>>(W2, W2T, kHid, kHid);
  k_cvt_transpose<<<(kHid * kD + 255) / 256, 256, 0, stream>>>(W3, W3T, kHid, kD);
  k_w1_pad_t<<<(kHid * kKPad + 255) / 256, 256, 0, stream>>>(W1, W1T);
  k_phi<<<(kMCP + 255) / 256, 256, 0, stream>>>(n2o, freqs, Phi);

  // E[0:1024] = U @ B
  k_gemm<0><<<dim3(kD / 128, kHot / 128), 256, 0, stream>>>(
      Ubf, BmT, Ebf, nullptr, kHot, kD, kRHot, kD,
      nullptr, nullptr, 0, nullptr, nullptr);
  // H1 = gelu(phi @ W1 + b1)
  k_gemm<0><<<dim3(kHid / 128, kMCP / 128), 256, 0, stream>>>(
      Phi, W1T, H1, nullptr, kMCP, kHid, kKPad, kHid,
      nullptr, b1, 1, nullptr, nullptr);
  // H2 = gelu(H1 @ W2 + b2)
  k_gemm<0><<<dim3(kHid / 128, kMCP / 128), 256, 0, stream>>>(
      H1, W2T, H2, nullptr, kMCP, kHid, kHid, kHid,
      nullptr, b2, 1, nullptr, nullptr);
  // E[1024:] = (H2 @ W3 + b3) * tail_gain
  k_gemm<0><<<dim3(kD / 128, kMCP / 128), 256, 0, stream>>>(
      H2, W3T, Ebf + (size_t)kHot * kD, nullptr, kMCP, kD, kHid, kD,
      nullptr, b3, 0, gain, nullptr);
  // latent = E[old_to_new[tokens]]
  k_latent<<<kTok, 256, 0, stream>>>(tokens, o2n, Ebf, Lat);
  // logits = (latent @ E[o2n[v]]^T) * exp(log_alpha) + out_bias
  int ntilesN = (kV + 127) / 128;   // 393
  k_gemm<1><<<dim3(ntilesN, kTok / 128), 256, 0, stream>>>(
      Lat, Ebf, nullptr, out, kTok, ntilesN * 128, kD, kV,
      o2n, obias, 0, nullptr, lalpha);
}
